// Encoder_25125558682008
// MI455X (gfx1250) — matrix operand order repaired and checked by reading, not compiled, not run on hardware
//
#include <hip/hip_runtime.h>

#define BN 256      // batch
#define NN 360      // nodes
#define NF 360      // features (== NN here)
#define NH 128      // hidden
#define KP 384      // K padded to multiple of 32 (360 -> 384)
#define MP 368      // h1 rows padded to multiple of 16
#define MT_FULL 24  // 384/16 m-tiles (producers of K-padded transposed operands)
#define MT_OUT  23  // 368/16 m-tiles (covers the 360 real rows)

typedef _Float16 h8_t __attribute__((ext_vector_type(8)));
typedef _Float16 v16h __attribute__((ext_vector_type(16)));
typedef float    v8f  __attribute__((ext_vector_type(8)));
typedef float    f4   __attribute__((ext_vector_type(4)));

static __device__ __forceinline__ v8f wmma16(v16h a, v16h b, v8f c) {
  // (neg_a, A, neg_b, B, c_mod, C, reuse_a, reuse_b)
  return __builtin_amdgcn_wmma_f32_16x16x32_f16(false, a, false, b, (short)0, c,
                                                false, false);
}

// load 8 consecutive f32 (two b128 loads) and convert to 8 f16
static __device__ __forceinline__ h8_t cvt8(const float* __restrict__ p) {
  const f4* q = (const f4*)p;
  f4 a = q[0];
  f4 b = q[1];
  h8_t r;
  r[0] = (_Float16)a[0]; r[1] = (_Float16)a[1];
  r[2] = (_Float16)a[2]; r[3] = (_Float16)a[3];
  r[4] = (_Float16)b[0]; r[5] = (_Float16)b[1];
  r[6] = (_Float16)b[2]; r[7] = (_Float16)b[3];
  return r;
}

static __device__ __forceinline__ v16h pack16(h8_t lo, h8_t hi) {
  v16h r;
#pragma unroll
  for (int i = 0; i < 8; ++i) { r[i] = lo[i]; r[8 + i] = hi[i]; }
  return r;
}

// B fragment: B^T stored row-major [N][Kpad]; lane n holds 16 f16
// at K = 8*half .. 8*half+7 and 16+8*half .. 16+8*half+7  (two 16B loads)
static __device__ __forceinline__ v16h loadB(const _Float16* __restrict__ p) {
  const h8_t* q = (const h8_t*)p;
  return pack16(q[0], q[2]);
}

// A fragment from an f32 row, K chunks at [s0, s0+8) and [s1, s1+8),
// zero-filled beyond the real K extent (chunks are 8-aligned: all-or-nothing).
static __device__ __forceinline__ v16h loadA_f32(const float* __restrict__ arow,
                                                 int s0, int s1) {
  h8_t c0 = {};
  h8_t c1 = {};
  if (s0 < NF) c0 = cvt8(arow + s0);
  if (s1 < NF) c1 = cvt8(arow + s1);
  return pack16(c0, c1);
}

// K-loop for GEMMs whose A operand is a row of an f32 [.,360] matrix (x or adj),
// converted to f16 in-flight; B^T is f16 with row stride KP.
// Software-pipelined: A fragment for step kt+1 is issued before the 8 WMMAs of
// step kt; all 8 B fragments get distinct registers so their 16 loads can be
// issued ahead and retired with partial loadcnt waits.
static __device__ __forceinline__ void gemm_f32A(const float* __restrict__ arow,
                                                 const _Float16* __restrict__ bt,
                                                 int nlo, int hf, v8f acc[8]) {
  const int a0 = 8 * hf;
  v16h a_cur = loadA_f32(arow, a0, a0 + 16);
#pragma unroll 1
  for (int kt = 0; kt < KP / 32; ++kt) {
    const int k0 = kt * 32;
    const _Float16* bp = bt + k0 + 8 * hf;
    v16h bf[8];
#pragma unroll
    for (int t = 0; t < 8; ++t)
      bf[t] = loadB(bp + (size_t)(t * 16 + nlo) * KP);
    v16h a_next = a_cur;
    if (kt + 1 < KP / 32) {
      const int s0 = k0 + 32 + a0;
      a_next = loadA_f32(arow, s0, s0 + 16);
    }
#pragma unroll
    for (int t = 0; t < 8; ++t)
      acc[t] = wmma16(a_cur, bf[t], acc[t]);
    a_cur = a_next;
  }
}

// ---------------- prep: transpose + pad weights to f16 ----------------
__global__ void prep_w1t(const float* __restrict__ W1, _Float16* __restrict__ w1t) {
  const int idx = blockIdx.x * blockDim.x + threadIdx.x;
  if (idx >= NH * KP) return;
  const int n = idx / KP, k = idx % KP;
  w1t[idx] = (k < NF) ? (_Float16)W1[(size_t)k * NH + n] : (_Float16)0.0f;
}

__global__ void prep_w2t(const float* __restrict__ W2, _Float16* __restrict__ w2t) {
  const int idx = blockIdx.x * blockDim.x + threadIdx.x;
  if (idx >= NH * NH) return;
  const int n = idx / NH, k = idx % NH;
  w2t[idx] = (_Float16)W2[(size_t)k * NH + n];
}

// ---------------- K1: XW1^T[b][n][m] = (x[b] @ W1)^T, f16, K-tail zeroed ----
__global__ __launch_bounds__(256) void k1_xw1(const float* __restrict__ x,
                                              const _Float16* __restrict__ w1t,
                                              _Float16* __restrict__ xw1t) {
  const int wid  = blockIdx.x * 8 + (threadIdx.x >> 5);
  const int lane = threadIdx.x & 31;
  const int nlo  = lane & 15;
  const int hf   = lane >> 4;
  const int b    = wid / MT_FULL;
  const int mt   = wid % MT_FULL;
  int row = mt * 16 + nlo;
  if (row >= NN) row = NN - 1;                       // clamp; masked on store
  const float* arow = x + ((size_t)b * NN + row) * NF;
  v8f zz = {};
  v8f acc[8];
#pragma unroll
  for (int t = 0; t < 8; ++t) acc[t] = zz;
  gemm_f32A(arow, w1t, nlo, hf, acc);
  const int mbase = mt * 16 + 8 * hf;                // 8 consecutive m per lane
#pragma unroll
  for (int t = 0; t < 8; ++t) {
    const int n = t * 16 + nlo;
    h8_t o;
#pragma unroll
    for (int r = 0; r < 8; ++r)
      o[r] = (mbase + r < NN) ? (_Float16)acc[t][r] : (_Float16)0.0f;
    *(h8_t*)(xw1t + ((size_t)b * NH + n) * KP + mbase) = o;   // 16B store
  }
}

// ---------------- K2: h1 = relu(adj @ XW1 + b1); partial column sums --------
__global__ __launch_bounds__(256) void k2_h1(const float* __restrict__ adj,
                                             const _Float16* __restrict__ xw1t,
                                             const float* __restrict__ b1,
                                             _Float16* __restrict__ h1,
                                             float* __restrict__ p1) {
  const int wid  = blockIdx.x * 8 + (threadIdx.x >> 5);
  const int lane = threadIdx.x & 31;
  const int nlo  = lane & 15;
  const int hf   = lane >> 4;
  const int b    = wid / MT_OUT;
  const int mt   = wid % MT_OUT;
  int row = mt * 16 + nlo;
  if (row >= NN) row = NN - 1;
  const float* arow = adj + ((size_t)b * NN + row) * NN;
  v8f zz = {};
  v8f acc[8];
#pragma unroll
  for (int t = 0; t < 8; ++t) acc[t] = zz;
  gemm_f32A(arow, xw1t + (size_t)b * NH * KP, nlo, hf, acc);
  const int mbase = mt * 16 + 8 * hf;
#pragma unroll
  for (int t = 0; t < 8; ++t) {
    const int n  = t * 16 + nlo;
    const float bv = b1[n];
    float cs = 0.0f;
#pragma unroll
    for (int r = 0; r < 8; ++r) {
      const int mm = mbase + r;                      // < MP always
      float v = acc[t][r] + bv;
      v = v > 0.0f ? v : 0.0f;
      h1[((size_t)b * MP + mm) * NH + n] = (_Float16)v;
      if (mm < NN) cs += v;                          // mask pad rows from sums
    }
    cs += __shfl_xor(cs, 16, 32);                    // fold rows 0-7 + 8-15
    if (hf == 0) p1[((size_t)b * MT_OUT + mt) * NH + n] = cs;
  }
}

// ---------------- K3: H1W2^T[b][n][m] = (h1 @ W2)^T, f16, K-tail zeroed -----
__global__ __launch_bounds__(256) void k3_h1w2(const _Float16* __restrict__ h1,
                                               const _Float16* __restrict__ w2t,
                                               _Float16* __restrict__ h1w2t) {
  const int wid  = blockIdx.x * 8 + (threadIdx.x >> 5);
  const int lane = threadIdx.x & 31;
  const int nlo  = lane & 15;
  const int hf   = lane >> 4;
  const int b    = wid / MT_FULL;
  const int mt   = wid % MT_FULL;
  int row = mt * 16 + nlo;
  if (row >= MP) row = MP - 1;
  const _Float16* arow = h1 + ((size_t)b * MP + row) * NH;
  v8f zz = {};
  v8f acc[8];
#pragma unroll
  for (int t = 0; t < 8; ++t) acc[t] = zz;
  {
    const h8_t* pa0 = (const h8_t*)(arow + 8 * hf);
    v16h a_cur = pack16(pa0[0], pa0[2]);             // chunks at +0 and +16 f16
#pragma unroll
    for (int kt = 0; kt < NH / 32; ++kt) {           // K = 128, no padding
      const int k0 = kt * 32;
      const _Float16* bp = w2t + k0 + 8 * hf;
      v16h bf[8];
#pragma unroll
      for (int t = 0; t < 8; ++t)
        bf[t] = loadB(bp + (size_t)(t * 16 + nlo) * NH);
      v16h a_next = a_cur;
      if (kt + 1 < NH / 32) {
        const h8_t* pa = (const h8_t*)(arow + k0 + 32 + 8 * hf);
        a_next = pack16(pa[0], pa[2]);
      }
#pragma unroll
      for (int t = 0; t < 8; ++t)
        acc[t] = wmma16(a_cur, bf[t], acc[t]);
      a_cur = a_next;
    }
  }
  const int mbase = mt * 16 + 8 * hf;
#pragma unroll
  for (int t = 0; t < 8; ++t) {
    const int n = t * 16 + nlo;
    h8_t o;
#pragma unroll
    for (int r = 0; r < 8; ++r)
      o[r] = (mbase + r < NN) ? (_Float16)acc[t][r] : (_Float16)0.0f;
    *(h8_t*)(h1w2t + ((size_t)b * NH + n) * KP + mbase) = o;
  }
}

// ---------------- K4: h2 = relu(adj @ H1W2 + b2) -> d_out; partial sums -----
__global__ __launch_bounds__(256) void k4_h2(const float* __restrict__ adj,
                                             const _Float16* __restrict__ h1w2t,
                                             const float* __restrict__ b2,
                                             float* __restrict__ out,
                                             float* __restrict__ p2) {
  const int wid  = blockIdx.x * 8 + (threadIdx.x >> 5);
  const int lane = threadIdx.x & 31;
  const int nlo  = lane & 15;
  const int hf   = lane >> 4;
  const int b    = wid / MT_OUT;
  const int mt   = wid % MT_OUT;
  int row = mt * 16 + nlo;
  if (row >= NN) row = NN - 1;
  const float* arow = adj + ((size_t)b * NN + row) * NN;
  v8f zz = {};
  v8f acc[8];
#pragma unroll
  for (int t = 0; t < 8; ++t) acc[t] = zz;
  gemm_f32A(arow, h1w2t + (size_t)b * NH * KP, nlo, hf, acc);
  const int mbase = mt * 16 + 8 * hf;
#pragma unroll
  for (int t = 0; t < 8; ++t) {
    const int n  = t * 16 + nlo;
    const float bv = b2[n];
    float cs = 0.0f;
#pragma unroll
    for (int r = 0; r < 8; ++r) {
      const int mm = mbase + r;
      float v = acc[t][r] + bv;
      v = v > 0.0f ? v : 0.0f;
      if (mm < NN) {
        out[((size_t)b * NN + mm) * NH + n] = v;
        cs += v;
      }
    }
    cs += __shfl_xor(cs, 16, 32);
    if (hf == 0) p2[((size_t)b * MT_OUT + mt) * NH + n] = cs;
  }
}

// ---------------- K5: deterministic reduction of partials -> gh -------------
__global__ void k5_gh(const float* __restrict__ p1, const float* __restrict__ p2,
                      float* __restrict__ gh) {
  const int idx = blockIdx.x * blockDim.x + threadIdx.x;
  if (idx >= BN * NH) return;
  const int b = idx / NH, n = idx % NH;
  float s1 = 0.0f, s2 = 0.0f;
  for (int t = 0; t < MT_OUT; ++t) {
    s1 += p1[((size_t)b * MT_OUT + t) * NH + n];
    s2 += p2[((size_t)b * MT_OUT + t) * NH + n];
  }
  gh[(size_t)b * 2 * NH + n]      = s1;
  gh[(size_t)b * 2 * NH + NH + n] = s2;
}

extern "C" void kernel_launch(void* const* d_in, const int* in_sizes, int n_in,
                              void* d_out, int out_size, void* d_ws, size_t ws_size,
                              hipStream_t stream) {
  (void)in_sizes; (void)n_in; (void)out_size; (void)ws_size;
  const float* x   = (const float*)d_in[0];
  const float* adj = (const float*)d_in[1];
  const float* W1  = (const float*)d_in[2];
  const float* b1  = (const float*)d_in[3];
  const float* W2  = (const float*)d_in[4];
  const float* b2  = (const float*)d_in[5];
  float* out = (float*)d_out;                     // h2: [B,360,128]
  float* gh  = out + (size_t)BN * NN * NH;        // gh: [B,256]

  char* ws = (char*)d_ws;
  size_t off = 0;
  auto alloc = [&](size_t bytes) -> void* {
    off = (off + 255) & ~(size_t)255;
    void* p = ws + off;
    off += bytes;
    return p;
  };
  _Float16* xw1t  = (_Float16*)alloc((size_t)BN * NH * KP * 2);  // XW1^T
  _Float16* h1    = (_Float16*)alloc((size_t)BN * MP * NH * 2);  // h1 row-major
  _Float16* h1w2t = (_Float16*)alloc((size_t)BN * NH * KP * 2);  // (h1@W2)^T
  _Float16* w1t   = (_Float16*)alloc((size_t)NH * KP * 2);
  _Float16* w2t   = (_Float16*)alloc((size_t)NH * NH * 2);
  float*    p1    = (float*)alloc((size_t)BN * MT_OUT * NH * 4);
  float*    p2    = (float*)alloc((size_t)BN * MT_OUT * NH * 4);

  prep_w1t<<<(NH * KP + 255) / 256, 256, 0, stream>>>(W1, w1t);
  prep_w2t<<<(NH * NH + 255) / 256, 256, 0, stream>>>(W2, w2t);
  k1_xw1 <<<BN * MT_FULL / 8, 256, 0, stream>>>(x, w1t, xw1t);
  k2_h1  <<<BN * MT_OUT  / 8, 256, 0, stream>>>(adj, xw1t, b1, h1, p1);
  k3_h1w2<<<BN * MT_FULL / 8, 256, 0, stream>>>(h1, w2t, h1w2t);
  k4_h2  <<<BN * MT_OUT  / 8, 256, 0, stream>>>(adj, h1w2t, b2, out, p2);
  k5_gh  <<<(BN * NH + 255) / 256, 256, 0, stream>>>(p1, p2, gh);
}
